// VQLinearPackTensorCore_6030134083829
// MI455X (gfx1250) — compile-verified, matrix operand order checked by reading, not run on hardware
//
#include <hip/hip_runtime.h>

typedef __attribute__((ext_vector_type(16))) _Float16 v16h;
typedef __attribute__((ext_vector_type(8)))  _Float16 v8h;
typedef __attribute__((ext_vector_type(8)))  float    v8f;
typedef __attribute__((ext_vector_type(4)))  unsigned long long v4ul;
typedef __attribute__((ext_vector_type(4)))  unsigned int v4u;
typedef __attribute__((ext_vector_type(8)))  int v8i_t;
typedef __attribute__((ext_vector_type(4)))  int v4i_t;

constexpr int IN_F   = 8192;   // in_features
constexpr int OUT_F  = 8192;   // out_features
constexpr int MROWS  = 16;     // batch rows
constexpr int WPR    = 512;    // int32 words per output row
constexpr int KCHUNK = 1024;   // K elements staged in LDS per chunk
constexpr int WAVES  = 8;      // waves per block (wave32)
constexpr int KSPLIT = 2;      // deterministic 2-way split-K
constexpr int KPER   = IN_F / KSPLIT;

__global__ __launch_bounds__(WAVES * 32)
void vq_wmma_kernel(const _Float16* __restrict__ inp,
                    const int*      __restrict__ qw,
                    const int*      __restrict__ lut32,   // 16 packed (f16,f16)
                    float*          __restrict__ ws)
{
  __shared__ _Float16 As[MROWS * KCHUNK];        // 32 KiB activation tile
  __shared__ unsigned long long byteTab[256];    // 2 KiB: byte -> 2 dequant pairs

  const int lane = threadIdx.x & 31;
  const int wave = threadIdx.x >> 5;
  const int half = lane >> 4;               // 0: lanes 0-15, 1: lanes 16-31
  const int col  = lane & 15;
  const int sh   = half << 4;

  // Build byte-decode table: entry b = { lut[b&15] , lut[b>>4] } as one b64.
  {
    const int b = threadIdx.x;              // exactly 256 threads
    const unsigned long long lo = (unsigned)lut32[b & 15];
    const unsigned long long hi = (unsigned)lut32[(b >> 4) & 15];
    byteTab[b] = lo | (hi << 32);
  }

  const int tile = blockIdx.x * WAVES + wave;   // 0..511
  const int n    = (tile << 4) + col;           // global output column
  const int kbeg = blockIdx.y * KPER;
  const int* qrow = qw + (size_t)n * WPR;

  v8f acc = {};

  for (int kc = kbeg; kc < kbeg + KPER; kc += KCHUNK) {
    __syncthreads();   // previous chunk consumed (also fences byteTab build)

    if (wave == 0) {
      // ---- Tensor Data Mover: stage A tile (KCHUNK x MROWS fp16) ----
      const unsigned lds = (unsigned)(uintptr_t)&As[0];       // LDS byte offset
      const unsigned long long ga = (unsigned long long)(uintptr_t)(inp + kc);
      // D# group0: count=1 | lds_addr | global_addr[56:0] | type=2
      v4u g0 = { 1u, lds, (unsigned)ga,
                 (unsigned)((ga >> 32) & 0x1FFFFFFu) | (2u << 30) };
      // D# group1: data_size=2B; tensor_dim0=IN_F, tensor_dim1=MROWS,
      //            tile_dim0=KCHUNK, tile_dim1=MROWS, dim0_stride=IN_F
      v8i_t g1 = { (int)(1u << 16),                          // data_size=1 (2B)
                   (int)(((unsigned)IN_F & 0xFFFFu) << 16),  // tensor_dim0[15:0]
                   (int)(((unsigned)MROWS) << 16),           // dim0[31:16]=0 | dim1[15:0]
                   (int)(((unsigned)KCHUNK) << 16),          // dim1[31:16]=0 | tile_dim0
                   MROWS,                                    // tile_dim1 | tile_dim2=0
                   IN_F,                                     // dim0_stride[31:0]
                   0, 0 };
      v4i_t gz4 = {0, 0, 0, 0};
      v8i_t gz8 = {0, 0, 0, 0, 0, 0, 0, 0};
      // amdgpu-toolchain (clang-23) 6-arg form
      __builtin_amdgcn_tensor_load_to_lds(g0, g1, gz4, gz4, gz8, 0);
      __builtin_amdgcn_s_wait_tensorcnt(0);
    }
    __syncthreads();

    for (int k0 = 0; k0 < KCHUNK; k0 += 64) {
      __builtin_prefetch(qrow + (((kc + k0) >> 4) + 128), 0, 1);
      const int4 q4 = *(const int4*)(qrow + ((kc + k0) >> 4));

      #pragma unroll
      for (int s = 0; s < 2; ++s) {          // two K=32 WMMA steps
        // ---- A fragment from LDS ----
        const int kb = k0 + (s << 5) + (half << 3);
        v8h alo = *(const v8h*)&As[col * KCHUNK + kb];
        v8h ahi = *(const v8h*)&As[col * KCHUNK + kb + 16];
        v16h a = __builtin_shufflevector(alo, ahi,
                   0,1,2,3,4,5,6,7,8,9,10,11,12,13,14,15);

        // ---- B fragment: 4 byte-table lookups (ds_load_b64 each) ----
        const unsigned w0 = ((unsigned)(s ? q4.z : q4.x)) >> sh;  // K=k..k+15 half
        const unsigned w1 = ((unsigned)(s ? q4.w : q4.y)) >> sh;  // K=k+16..+31 half
        v4ul bl = { byteTab[w0 & 0xFFu], byteTab[(w0 >> 8) & 0xFFu],
                    byteTab[w1 & 0xFFu], byteTab[(w1 >> 8) & 0xFFu] };
        v16h b = __builtin_bit_cast(v16h, bl);

        acc = __builtin_amdgcn_wmma_f32_16x16x32_f16(
                false, a, false, b, (short)0, acc, false, false);
      }
    }
  }

  // D layout (f32 16x16): VGPR r -> row r + 8*half, lane%16 -> column.
  #pragma unroll
  for (int r = 0; r < 8; ++r) {
    const int m = r + (half << 3);
    atomicAdd(ws + (size_t)m * OUT_F + n, acc[r]);   // global_atomic_add_f32
  }
}

__global__ void vq_finalize(const float* __restrict__ ws,
                            _Float16* __restrict__ out, int nElem)
{
  int i = blockIdx.x * blockDim.x + threadIdx.x;
  if (i < nElem) out[i] = (_Float16)ws[i];
}

extern "C" void kernel_launch(void* const* d_in, const int* in_sizes, int n_in,
                              void* d_out, int out_size, void* d_ws, size_t ws_size,
                              hipStream_t stream) {
  const _Float16* inp = (const _Float16*)d_in[0];   // (1,16,8192) fp16
  const int*      qw  = (const int*)d_in[1];        // (8192,512) int32
  const int*      lut = (const int*)d_in[2];        // (16,2) fp16 -> 16 dwords
  float*    ws  = (float*)d_ws;                     // 16x8192 f32 accumulator
  _Float16* out = (_Float16*)d_out;                 // (1,16,8192) fp16

  (void)hipMemsetAsync(ws, 0, (size_t)MROWS * OUT_F * sizeof(float), stream);

  dim3 grid(OUT_F / (16 * WAVES), KSPLIT);
  vq_wmma_kernel<<<grid, WAVES * 32, 0, stream>>>(inp, qw, lut, ws);

  const int nE = MROWS * OUT_F;
  vq_finalize<<<(nE + 255) / 256, 256, 0, stream>>>(ws, out, nE);
}
